// DPNModel_63977832841520
// MI455X (gfx1250) — matrix operand order repaired and checked by reading, not compiled, not run on hardware
//
#include <hip/hip_runtime.h>
#include <hip/hip_bf16.h>

// ---------------------------------------------------------------------------
// DPN forward for MI455X (gfx1250, wave32, WMMA + TDM).
//
// f16 implicit-GEMM convs: 4 waves/block, 64 chans x 32 pixels per block.
//   Per 64-K stage: TDM loads the 64x64 A-tile (padded f16 weights) to LDS,
//   128 threads build the 32x64 im2col B-chunk in LDS, each wave issues
//   4x v_wmma_f32_16x16x32_f16 (2 K-slices x 2 N-tiles, A reused twice).
// conv14 (512->4096 7x7, 411MB f32 weights > L2): V_WMMA_F32_16X16X4_F32
//   reading f32 weights directly (casting would double HBM traffic).
// local filtering (484MB single-use weights): streaming block reduce.
// ---------------------------------------------------------------------------

typedef __attribute__((ext_vector_type(16))) _Float16 v16h;
typedef __attribute__((ext_vector_type(8)))  _Float16 v8h;
typedef __attribute__((ext_vector_type(8)))  float    v8f;
typedef __attribute__((ext_vector_type(2)))  float    v2f;
typedef unsigned int v4u __attribute__((ext_vector_type(4)));
typedef int          v4i __attribute__((ext_vector_type(4)));
typedef int          v8i __attribute__((ext_vector_type(8)));

#define CC   21
#define HH   48
#define WW   48
#define MM   5
#define KKF  50
#define NPIX (HH * WW)

#if __has_builtin(__builtin_amdgcn_tensor_load_to_lds) && \
    __has_builtin(__builtin_amdgcn_s_wait_tensorcnt)
#define HAVE_TDM 1
#if __has_include(<hip/amd_detail/amd_gfx1250_TDM.h>)
#define TDM_6ARG 1   // therock-10.0 headers -> clang-23 6-arg builtin
#warning "CDNA5 probe: TDM path ENABLED (6-arg tensor_load_to_lds)"
#else
#define TDM_6ARG 0   // ROCm 7.2 clang-22 5-arg builtin
#warning "CDNA5 probe: TDM path ENABLED (5-arg tensor_load_to_lds)"
#endif
#else
#define HAVE_TDM 0
#warning "CDNA5 probe: TDM builtin ABSENT - cooperative-copy fallback"
#endif

#if __has_builtin(__builtin_amdgcn_wmma_f32_16x16x4_f32)
#define HAVE_WMMA_F32K4 1
#warning "CDNA5 probe: wmma_f32_16x16x4_f32 AVAILABLE"
#else
#define HAVE_WMMA_F32K4 0
#warning "CDNA5 probe: wmma_f32_16x16x4_f32 ABSENT - scalar fallback"
#endif

// ---------------------------------------------------------------------------
// f32 -> f16 cast into zero-padded [Mp][Kp] weight layout
// ---------------------------------------------------------------------------
__global__ void cast_pad_f16(const float* __restrict__ src,
                             _Float16* __restrict__ dst,
                             int Cout, int Kdim, int Mp, int Kp) {
  int i = blockIdx.x * blockDim.x + threadIdx.x;
  if (i >= Mp * Kp) return;
  int m = i / Kp, k = i - m * Kp;
  dst[i] = (m < Cout && k < Kdim) ? (_Float16)src[(size_t)m * Kdim + k]
                                  : (_Float16)0.f;
}

__global__ void cast_f32_f16(const float* __restrict__ src,
                             _Float16* __restrict__ dst, int n) {
  int i = blockIdx.x * blockDim.x + threadIdx.x;
  if (i < n) dst[i] = (_Float16)src[i];
}

// ---------------------------------------------------------------------------
// im2col gathers (zero pad, dilation, stride 1)
// ---------------------------------------------------------------------------
__device__ __forceinline__ _Float16 im2col_h(const _Float16* __restrict__ Xi,
                                             int kb, int Kdim, int kk, int kk2,
                                             int dil, int pad, int oy, int ox,
                                             int Hh, int Ww, bool nvalid) {
  if (!nvalid || kb >= Kdim) return (_Float16)0.f;
  int cin = kb / kk2;  int r  = kb - cin * kk2;
  int kh  = r / kk;    int kw = r - kh * kk;
  int iy = oy - pad + kh * dil;
  int ix = ox - pad + kw * dil;
  if (iy < 0 || iy >= Hh || ix < 0 || ix >= Ww) return (_Float16)0.f;
  return Xi[(size_t)(cin * Hh + iy) * Ww + ix];
}

__device__ __forceinline__ float im2col_f(const float* __restrict__ Xi,
                                          int kb, int Kdim, int kk, int kk2,
                                          int dil, int pad, int oy, int ox,
                                          int Hh, int Ww, bool nvalid) {
  if (!nvalid || kb >= Kdim) return 0.f;
  int cin = kb / kk2;  int r  = kb - cin * kk2;
  int kh  = r / kk;    int kw = r - kh * kk;
  int iy = oy - pad + kh * dil;
  int ix = ox - pad + kw * dil;
  if (iy < 0 || iy >= Hh || ix < 0 || ix >= Ww) return 0.f;
  return Xi[(size_t)(cin * Hh + iy) * Ww + ix];
}

// ---------------------------------------------------------------------------
// f16 implicit-GEMM conv. Block = 128 threads = 4 waves:
//   64 output channels x 32 pixels, K staged 64 wide.
//   Per stage: TDM A-tile (64x64 f16) -> LDS, cooperative im2col B (32x64)
//   -> LDS, then each wave: 2 K-slices x 2 N-tiles = 4 WMMA, A reused 2x.
//   EXEC is all-1s at every WMMA (no early exits; uniform control flow).
// ---------------------------------------------------------------------------
__global__ void __launch_bounds__(128)
conv_wmma_f16(const _Float16* __restrict__ Wt,   // [Mp][Kp] padded
              const _Float16* __restrict__ Xi,   // [Cin][H][W]
              const float* __restrict__ bias,
              float* __restrict__ Yo,            // [Cout][H][W] f32
              _Float16* __restrict__ Yh,         // optional f16 copy
              int Cout, int Cin, int Hh, int Ww,
              int kk, int dil, int pad, int relu, int Kp) {
  const int tid  = threadIdx.x;
  const int wid  = tid >> 5;
  const int lane = tid & 31;
  const int g = lane >> 4, l = lane & 15;
  const int Npix = Hh * Ww;
  const int kk2  = kk * kk;
  const int Kdim = Cin * kk2;

  const int n0    = blockIdx.x * 32 + l;   // first pixel-tile column
  const int n1    = n0 + 16;               // second pixel-tile column
  const int mrow0 = blockIdx.y * 64;

  __shared__ _Float16 lds_a[64][64];       // A tile (TDM target), 8KB
  __shared__ _Float16 lds_b[32][64];       // B tile [pixel][K], 4KB

  v8f acc0 = {}, acc1 = {};
  for (int k0 = 0; k0 < Kp; k0 += 64) {
    __syncthreads();                       // previous stage fully consumed

#if HAVE_TDM
    if (wid == 0) {
      // ---- Tensor DMA descriptor (ISA 8.3/8.4), 2D tile 64x64, f16 ----
      unsigned int ldsaddr = (unsigned int)(uintptr_t)&lds_a[0][0];
      unsigned long long ga =
          (unsigned long long)(uintptr_t)(Wt + (size_t)mrow0 * Kp + k0);
      v4u g0;
      g0[0] = 1u;                                   // count=1, user mode
      g0[1] = ldsaddr;                              // lds_addr
      g0[2] = (unsigned int)ga;                     // global_addr[31:0]
      g0[3] = (unsigned int)((ga >> 32) & 0x1FFFFFFu) | (2u << 30); // type=2
      v8i g1;
      g1[0] = (1 << 16);                            // data_size = 2 bytes
      g1[1] = (int)(((unsigned)Kp & 0xFFFFu) << 16);        // tensor_dim0 lo
      g1[2] = (int)((((unsigned)Kp >> 16) & 0xFFFFu) |
                    (0xFFFFu << 16));               // dim0 hi | tensor_dim1 lo
      g1[3] = (int)(0x00FFu | (64u << 16));         // dim1 hi | tile_dim0=64
      g1[4] = 64;                                   // tile_dim1=64, tile_dim2=0
      g1[5] = Kp;                                   // tensor_dim0_stride lo
      g1[6] = 0;                                    // stride hi | dim1_stride lo
      g1[7] = 0;
      v4i g2 = {0, 0, 0, 0};                        // 2D: groups 2/3 unused
      v4i g3 = {0, 0, 0, 0};
#if TDM_6ARG
      v8i g4 = {0, 0, 0, 0, 0, 0, 0, 0};
      __builtin_amdgcn_tensor_load_to_lds(g0, g1, g2, g3, g4, 0);
#else
      __builtin_amdgcn_tensor_load_to_lds(g0, g1, g2, g3, 0);
#endif
      __builtin_amdgcn_s_wait_tensorcnt((short)0);
    }
#else
    // fallback: cooperative A-tile copy (aligned 32-bit loads)
    for (int i = tid; i < 64 * 32; i += 128) {
      int r = i >> 5, c2 = i & 31;
      ((unsigned int*)&lds_a[0][0])[r * 32 + c2] =
          *(const unsigned int*)(Wt + ((size_t)(mrow0 + r) * Kp + k0) + 2 * c2);
    }
#endif

    // ---- stage B chunk: 2048 halves, 16 per thread ----
    for (int e = tid; e < 2048; e += 128) {
      int bn = e >> 6, bk = e & 63;
      int nb = blockIdx.x * 32 + bn;
      bool nbv = nb < Npix;
      int nc = nbv ? nb : 0;
      int oyb = nc / Ww, oxb = nc - oyb * Ww;
      lds_b[bn][bk] =
          im2col_h(Xi, k0 + bk, Kdim, kk, kk2, dil, pad, oyb, oxb, Hh, Ww, nbv);
    }
    __syncthreads();

    // ---- 2 K-slices x 2 N-tiles, A fragment reused twice ----
#pragma unroll
    for (int ks = 0; ks < 2; ++ks) {
      const v8h* arow = (const v8h*)&lds_a[wid * 16 + l][ks * 32];
      v16h a = __builtin_shufflevector(arow[2 * g], arow[2 * g + 1],
                                       0, 1, 2, 3, 4, 5, 6, 7,
                                       8, 9, 10, 11, 12, 13, 14, 15);
      const v8h* b0r = (const v8h*)&lds_b[l][ks * 32];
      v16h b0 = __builtin_shufflevector(b0r[2 * g], b0r[2 * g + 1],
                                        0, 1, 2, 3, 4, 5, 6, 7,
                                        8, 9, 10, 11, 12, 13, 14, 15);
      const v8h* b1r = (const v8h*)&lds_b[16 + l][ks * 32];
      v16h b1 = __builtin_shufflevector(b1r[2 * g], b1r[2 * g + 1],
                                        0, 1, 2, 3, 4, 5, 6, 7,
                                        8, 9, 10, 11, 12, 13, 14, 15);
      acc0 = __builtin_amdgcn_wmma_f32_16x16x32_f16(
          false, a, false, b0, (short)0, acc0, false, false);
      acc1 = __builtin_amdgcn_wmma_f32_16x16x32_f16(
          false, a, false, b1, (short)0, acc1, false, false);
    }
  }

#pragma unroll
  for (int v = 0; v < 8; ++v) {
    int m = mrow0 + wid * 16 + 8 * g + v;        // C/D: M = 8g + v
    if (m < Cout) {
      float bv = bias[m];
      if (n0 < Npix) {
        float val = acc0[v] + bv;
        if (relu) val = fmaxf(val, 0.f);
        size_t o = (size_t)m * Npix + n0;
        Yo[o] = val;
        if (Yh) Yh[o] = (_Float16)val;
      }
      if (n1 < Npix) {
        float val = acc1[v] + bv;
        if (relu) val = fmaxf(val, 0.f);
        size_t o = (size_t)m * Npix + n1;
        Yo[o] = val;
        if (Yh) Yh[o] = (_Float16)val;
      }
    }
  }
}

// ---------------------------------------------------------------------------
// f32 implicit-GEMM conv via V_WMMA_F32_16X16X4_F32 for conv14: weights are
// read once from HBM in f32 (411MB > L2); im2col B staged in LDS per block.
// Requires Kdim % 32 == 0 (holds: 512*49 = 25088).
// ---------------------------------------------------------------------------
__global__ void __launch_bounds__(128)
conv_wmma_f32k4(const float* __restrict__ Wt, const float* __restrict__ Xi,
                const float* __restrict__ bias, float* __restrict__ Yo,
                _Float16* __restrict__ Yh,
                int Cout, int Cin, int Hh, int Ww,
                int kk, int dil, int pad, int relu) {
  const int tid  = threadIdx.x;
  const int wid  = tid >> 5;
  const int lane = tid & 31;
  const int g = lane >> 4, l = lane & 15;
  const int Npix = Hh * Ww;
  const int kk2  = kk * kk;
  const int Kdim = Cin * kk2;

  const int  n      = blockIdx.x * 16 + l;
  const bool nvalid = n < Npix;

  const int  ma     = blockIdx.y * 64 + wid * 16 + l;
  const bool mvalid = ma < Cout;
  const float* wrow = Wt + (size_t)(mvalid ? ma : 0) * Kdim;

  __shared__ float lds_bf[16][32];

  v8f acc = {};
  for (int k0 = 0; k0 < Kdim; k0 += 32) {
    __syncthreads();
    for (int e = tid; e < 512; e += 128) {
      int bn = e >> 5, bk = e & 31;
      int nb = blockIdx.x * 16 + bn;
      bool nbv = nb < Npix;
      int nc = nbv ? nb : 0;
      int oyb = nc / Ww, oxb = nc - oyb * Ww;
      lds_bf[bn][bk] =
          im2col_f(Xi, k0 + bk, Kdim, kk, kk2, dil, pad, oyb, oxb, Hh, Ww, nbv);
    }
    __syncthreads();

    if (mvalid && (k0 + 96) < Kdim)        // global_prefetch_b8 weight stream
      __builtin_prefetch((const void*)(wrow + k0 + 64), 0, 3);

#if HAVE_WMMA_F32K4
#pragma unroll
    for (int s = 0; s < 8; ++s) {
      int kr = s * 4;
      v2f a = {0.f, 0.f};
      if (mvalid) a = *(const v2f*)(wrow + k0 + kr + 2 * g);  // A/B: K = 2g+v
      v2f b = *(const v2f*)&lds_bf[l][kr + 2 * g];
      acc = __builtin_amdgcn_wmma_f32_16x16x4_f32(
          false, a, false, b, (short)0, acc, false, false);
    }
#else
    for (int kr = 0; kr < 32; ++kr) {       // correct layout-free fallback
      float bv = lds_bf[l][kr];
#pragma unroll
      for (int v = 0; v < 8; ++v) {
        int m = blockIdx.y * 64 + wid * 16 + 8 * g + v;
        float av = (m < Cout) ? Wt[(size_t)m * Kdim + k0 + kr] : 0.f;
        acc[v] += av * bv;
      }
    }
#endif
  }

#pragma unroll
  for (int v = 0; v < 8; ++v) {
    int m = blockIdx.y * 64 + wid * 16 + 8 * g + v;
    if (nvalid && m < Cout) {
      float val = acc[v] + bias[m];
      if (relu) val = fmaxf(val, 0.f);
      size_t o = (size_t)m * Npix + n;
      Yo[o] = val;
      if (Yh) Yh[o] = (_Float16)val;
    }
  }
}

// ---------------------------------------------------------------------------
// 2x2 maxpool (f32 + f16 outputs)
// ---------------------------------------------------------------------------
__global__ void maxpool2(const float* __restrict__ in, float* __restrict__ out,
                         _Float16* __restrict__ outh, int Cc, int Hi, int Wi) {
  int Ho = Hi >> 1, Wo = Wi >> 1;
  int tot = Cc * Ho * Wo;
  int idx = blockIdx.x * blockDim.x + threadIdx.x;
  if (idx >= tot) return;
  int c = idx / (Ho * Wo); int r = idx - c * Ho * Wo;
  int y = r / Wo, x = r - y * Wo;
  const float* p = in + ((size_t)c * Hi + 2 * y) * Wi + 2 * x;
  float v = fmaxf(fmaxf(p[0], p[1]), fmaxf(p[Wi], p[Wi + 1]));
  out[idx]  = v;
  outh[idx] = (_Float16)v;
}

// ---------------------------------------------------------------------------
// bilinear upsample 6x6 -> 48x48 (align-corners linspace) + sigmoid
// ---------------------------------------------------------------------------
__global__ void upsample_sigmoid(const float* __restrict__ in,
                                 float* __restrict__ out) {
  int idx = blockIdx.x * blockDim.x + threadIdx.x;
  if (idx >= CC * NPIX) return;
  int c = idx / NPIX; int r = idx - c * NPIX;
  int y = r / WW, x = r - y * WW;
  float fy = (float)y * 5.0f / 47.0f;
  float fx = (float)x * 5.0f / 47.0f;
  int y0 = (int)floorf(fy); int y1 = min(y0 + 1, 5);
  int x0 = (int)floorf(fx); int x1 = min(x0 + 1, 5);
  float wy = fy - (float)y0, wx = fx - (float)x0;
  const float* pc = in + c * 36;
  float v00 = pc[y0 * 6 + x0], v01 = pc[y0 * 6 + x1];
  float v10 = pc[y1 * 6 + x0], v11 = pc[y1 * 6 + x1];
  float r0 = v00 * (1.f - wy) + v10 * wy;
  float r1 = v01 * (1.f - wy) + v11 * wy;
  float v  = r0 * (1.f - wx) + r1 * wx;
  out[idx] = 1.f / (1.f + expf(-v));
}

// ---------------------------------------------------------------------------
// Per-pixel 50x50 local filter: one block per output element, 256 threads
// stream the 2500-weight run coalesced and tree-reduce in LDS.
// 484MB of weights used once each -> pure-bandwidth kernel.
// ---------------------------------------------------------------------------
__global__ void __launch_bounds__(256)
local_filter(const float* __restrict__ unary, const float* __restrict__ lw,
             const float* __restrict__ lb, float* __restrict__ out,
             _Float16* __restrict__ outh) {
  int p = blockIdx.x;
  int c = p / NPIX; int yx = p - c * NPIX;
  int y = yx / WW, x = yx - y * WW;
  const float* wrow = lw + (size_t)p * (KKF * KKF);
  float s = 0.f;
  for (int i = threadIdx.x; i < KKF * KKF; i += blockDim.x) {
    int u = i / KKF, v = i - u * KKF;
    int iy = y + u - 25, ix = x + v - 25;   // PAD = (25, 24)
    float uv = 0.f;
    if (iy >= 0 && iy < HH && ix >= 0 && ix < WW)
      uv = unary[c * NPIX + iy * WW + ix];
    s += uv * wrow[i];
  }
  __shared__ float red[256];
  red[threadIdx.x] = s;
  __syncthreads();
  for (int st = 128; st > 0; st >>= 1) {
    if ((int)threadIdx.x < st) red[threadIdx.x] += red[threadIdx.x + st];
    __syncthreads();
  }
  if (threadIdx.x == 0) {
    float v = red[0] + lb[p];
    out[p]  = v;
    outh[p] = (_Float16)v;
  }
}

// ---------------------------------------------------------------------------
// smooth = min over M groups; out = softmax_c(log(unary) - smooth)
// ---------------------------------------------------------------------------
__global__ void smooth_softmax(const float* __restrict__ unary,
                               const float* __restrict__ g2,
                               float* __restrict__ out) {
  int p = blockIdx.x * blockDim.x + threadIdx.x;
  if (p >= NPIX) return;
  float z[CC];
  float mx = -1e30f;
#pragma unroll
  for (int c = 0; c < CC; ++c) {
    float sm = g2[(size_t)c * NPIX + p];
#pragma unroll
    for (int m = 1; m < MM; ++m)
      sm = fminf(sm, g2[(size_t)(m * CC + c) * NPIX + p]);
    float v = logf(unary[(size_t)c * NPIX + p]) - sm;
    z[c] = v;
    mx = fmaxf(mx, v);
  }
  float s = 0.f;
#pragma unroll
  for (int c = 0; c < CC; ++c) { z[c] = expf(z[c] - mx); s += z[c]; }
  float inv = 1.f / s;
#pragma unroll
  for (int c = 0; c < CC; ++c) out[(size_t)c * NPIX + p] = z[c] * inv;
}

// ---------------------------------------------------------------------------
// host orchestration
// ---------------------------------------------------------------------------
extern "C" void kernel_launch(void* const* d_in, const int* in_sizes, int n_in,
                              void* d_out, int out_size, void* d_ws, size_t ws_size,
                              hipStream_t stream) {
  (void)in_sizes; (void)n_in; (void)out_size; (void)ws_size;

  const float* x = (const float*)d_in[0];
  const float* vw[15]; const float* vb[15];
  for (int i = 0; i < 15; ++i) {
    vw[i] = (const float*)d_in[1 + i];
    vb[i] = (const float*)d_in[16 + i];
  }
  const float* local_w = (const float*)d_in[31];
  const float* local_b = (const float*)d_in[32];
  const float* lin1_w  = (const float*)d_in[33];
  const float* lin1_b  = (const float*)d_in[34];
  const float* glob_w  = (const float*)d_in[35];
  const float* glob_b  = (const float*)d_in[36];
  const float* lin2_w  = (const float*)d_in[37];
  const float* lin2_b  = (const float*)d_in[38];
  float* out = (float*)d_out;

  char* p = (char*)d_ws;
  auto alloc = [&](size_t bytes) -> char* {
    char* r = p; p += (bytes + 255) & ~(size_t)255; return r;
  };
  float*    act32[2] = {(float*)alloc(160 * 1024 * 4), (float*)alloc(160 * 1024 * 4)};
  _Float16* act16[2] = {(_Float16*)alloc(160 * 1024 * 2), (_Float16*)alloc(160 * 1024 * 2)};
  _Float16* w16   = (_Float16*)alloc(2400000 * 2);   // max padded layer: 512*4608
  float*    unary = (float*)alloc(CC * NPIX * 4);
  float*    loc32 = (float*)alloc(CC * NPIX * 4);
  _Float16* loc16 = (_Float16*)alloc(CC * NPIX * 2);
  float*    t32   = (float*)alloc(CC * NPIX * 4);
  _Float16* t16   = (_Float16*)alloc(CC * NPIX * 2);
  float*    g32   = (float*)alloc(CC * MM * NPIX * 4);
  _Float16* g16   = (_Float16*)alloc(CC * MM * NPIX * 2);
  float*    g2    = (float*)alloc(CC * MM * NPIX * 4);

  cast_f32_f16<<<(3 * NPIX + 255) / 256, 256, 0, stream>>>(x, act16[0], 3 * NPIX);

  struct LCfg { int cin, cout, k, d, pd, h; };
  const LCfg cfg[15] = {
    {3, 64, 3, 1, 1, 48},   {64, 64, 3, 1, 1, 48},
    {64, 128, 3, 1, 1, 24}, {128, 128, 3, 1, 1, 24},
    {128, 256, 3, 1, 1, 12},{256, 256, 3, 1, 1, 12},{256, 256, 3, 1, 1, 12},
    {256, 512, 3, 1, 1, 6}, {512, 512, 3, 1, 1, 6}, {512, 512, 3, 1, 1, 6},
    {512, 512, 3, 2, 2, 6}, {512, 512, 3, 2, 2, 6}, {512, 512, 3, 2, 2, 6},
    {512, 4096, 7, 4, 12, 6},{4096, 21, 1, 1, 0, 6}};
  const bool poolAfter[15] = {false, true, false, true, false, false, true,
                              false, false, false, false, false, false, false, false};

  auto launch_f16_conv = [&](const float* wsrc, const _Float16* xin,
                             const float* bsrc, float* yo, _Float16* yh,
                             int Cout, int Cin, int h, int k, int d, int pd,
                             int relu) {
    int Kdim = Cin * k * k;
    int Kp = (Kdim + 63) & ~63;        // K padded to 64 (TDM tile width)
    int Mp = (Cout + 63) & ~63;        // M padded to 64 (TDM tile height)
    cast_pad_f16<<<(Mp * Kp + 255) / 256, 256, 0, stream>>>(wsrc, w16, Cout,
                                                            Kdim, Mp, Kp);
    dim3 grid((h * h + 31) / 32, Mp / 64);
    conv_wmma_f16<<<grid, 128, 0, stream>>>(w16, xin, bsrc, yo, yh,
                                            Cout, Cin, h, h, k, d, pd, relu, Kp);
  };

  int cur = 0;
  for (int li = 0; li < 15; ++li) {
    const LCfg& c = cfg[li];
    int nxt = cur ^ 1;
    if (li == 13) {
      dim3 grid((c.h * c.h + 15) / 16, (c.cout + 63) / 64);
      conv_wmma_f32k4<<<grid, 128, 0, stream>>>(
          vw[li], act32[cur], vb[li], act32[nxt], act16[nxt],
          c.cout, c.cin, c.h, c.h, c.k, c.d, c.pd, 1);
    } else {
      launch_f16_conv(vw[li], act16[cur], vb[li], act32[nxt], act16[nxt],
                      c.cout, c.cin, c.h, c.k, c.d, c.pd, 1);
    }
    cur = nxt;
    if (poolAfter[li]) {
      int Ho = c.h / 2, tot = c.cout * Ho * Ho;
      nxt = cur ^ 1;
      maxpool2<<<(tot + 255) / 256, 256, 0, stream>>>(
          act32[cur], act32[nxt], act16[nxt], c.cout, c.h, c.h);
      cur = nxt;
    }
  }

  upsample_sigmoid<<<(CC * NPIX + 255) / 256, 256, 0, stream>>>(act32[cur], unary);
  local_filter<<<CC * NPIX, 256, 0, stream>>>(unary, local_w, local_b, loc32, loc16);

  // lin1 (C x C) as 1x1 conv
  launch_f16_conv(lin1_w, loc16, lin1_b, t32, t16, CC, CC, HH, 1, 1, 0, 0);
  // glob conv 21 -> 105, 9x9 pad 4
  launch_f16_conv(glob_w, t16, glob_b, g32, g16, CC * MM, CC, HH, 9, 1, 4, 0);
  // lin2 (105 x 105) as 1x1 conv
  launch_f16_conv(lin2_w, g16, lin2_b, g2, (_Float16*)nullptr,
                  CC * MM, CC * MM, HH, 1, 1, 0, 0);

  smooth_softmax<<<(NPIX + 255) / 256, 256, 0, stream>>>(unary, g2, out);
}
